// MetaNARM_80762565034079
// MI455X (gfx1250) — compile-verified, matrix operand order checked by reading, not run on hardware
//
#include <hip/hip_runtime.h>

// ---------------------------------------------------------------------------
// Shapes (fixed by the reference)
// ---------------------------------------------------------------------------
#define BATCH   1024
#define T1_LEN  198
#define T_FULL  199          // T1 + T2
#define DIM     256          // D == H == 256
#define M_ROWS  (BATCH * T_FULL)   // 203776, divisible by 16

typedef __attribute__((ext_vector_type(16))) __bf16          v16bf;
typedef __attribute__((ext_vector_type(8)))  float           v8f;
typedef __attribute__((ext_vector_type(8)))  unsigned short  u16x8;

#define WMMA_BF16(A, B, C) \
  __builtin_amdgcn_wmma_f32_16x16x32_bf16(false, (A), false, (B), (short)0, (C), false, false)

// ---------------------------------------------------------------------------
// bf16 helpers (round-to-nearest-even)
// ---------------------------------------------------------------------------
__device__ __forceinline__ unsigned short f2bf_bits(float f) {
  unsigned u = __builtin_bit_cast(unsigned, f);
  u += 0x7FFFu + ((u >> 16) & 1u);
  return (unsigned short)(u >> 16);
}
__device__ __forceinline__ __bf16 bits2bf(unsigned short h) {
  __bf16 r;
  __builtin_memcpy(&r, &h, 2);
  return r;
}
__device__ __forceinline__ float bfbits2f(unsigned short h) {
  unsigned u = ((unsigned)h) << 16;
  return __builtin_bit_cast(float, u);
}
__device__ __forceinline__ float sigf(float x) { return 1.0f / (1.0f + __expf(-x)); }

// ---------------------------------------------------------------------------
// WMMA 16x16x32 bf16 operand tile loader.
// ISA 16-bit operand striping (CDNA5 ISA §7.12.2):
//   lanes  0-15 : row = lane,      K-chunks {kb+0..7,  kb+16..23}
//   lanes 16-31 : row = lane-16,   K-chunks {kb+8..15, kb+24..31}
// Caller resolves the per-lane row pointer (row index = lane & 15); this works
// for both the A matrix (rows of the left operand) and B = Wᵀ (rows of W,
// i.e. output columns), since W is stored row-major [N, K].
// ---------------------------------------------------------------------------
__device__ __forceinline__ v16bf load_tile_bf16(const unsigned short* __restrict__ row,
                                                int kb, int lane) {
  const int half = (lane >> 4) & 1;
  u16x8 a = *reinterpret_cast<const u16x8*>(row + kb + half * 8);
  u16x8 b = *reinterpret_cast<const u16x8*>(row + kb + 16 + half * 8);
  v16bf r;
#pragma unroll
  for (int i = 0; i < 8; ++i) { r[i] = bits2bf(a[i]); r[8 + i] = bits2bf(b[i]); }
  return r;
}

// ---------------------------------------------------------------------------
// Kernel 0: f32 -> bf16 bulk conversion (weights + embedding table, once/call)
// ---------------------------------------------------------------------------
__global__ void cvt_f32_bf16_kernel(const float* __restrict__ src,
                                    unsigned short* __restrict__ dst, int n) {
  int i = blockIdx.x * blockDim.x + threadIdx.x;
  if (i < n) dst[i] = f2bf_bits(src[i]);
}

// ---------------------------------------------------------------------------
// Kernel 1: zero-init GRU state
// ---------------------------------------------------------------------------
__global__ void init_state_kernel(float* __restrict__ h_f32,
                                  unsigned short* __restrict__ h_bf) {
  int i = blockIdx.x * blockDim.x + threadIdx.x;
  if (i < BATCH * DIM) { h_f32[i] = 0.0f; h_bf[i] = 0; }
}

// ---------------------------------------------------------------------------
// Kernel 2: one fused GRU timestep.
// Each 1-wave block owns a 16(batch) x 16(hidden) tile and computes all
// 6 gate projections (xr,xz,xn,hr,hz,hn) with WMMA, then the gate math.
// h bf16 is ping-ponged: other blocks read the full h row during their GEMM,
// so writes go to the other buffer.
// ---------------------------------------------------------------------------
__global__ void gru_step_kernel(const int* __restrict__ seq1,
                                const int* __restrict__ seq2,
                                const unsigned short* __restrict__ emb_bf,
                                const unsigned short* __restrict__ wih_bf,
                                const unsigned short* __restrict__ whh_bf,
                                const float* __restrict__ b_ih,
                                const float* __restrict__ b_hh,
                                float* __restrict__ h_f32,
                                const unsigned short* __restrict__ h_rd,
                                unsigned short* __restrict__ h_wr,
                                unsigned short* __restrict__ gru_bf,
                                int t) {
  const int lane = threadIdx.x;
  const int b0 = blockIdx.x * 16;
  const int n0 = blockIdx.y * 16;
  const int r15 = lane & 15;
  const int half = lane >> 4;

  // Per-lane A rows: embedding row of token (b0+r15, t) and h row (b0+r15)
  const int arow = b0 + r15;
  const int tok = (t < T1_LEN) ? seq1[arow * T1_LEN + t] : seq2[arow];
  const unsigned short* xrow = emb_bf + (size_t)tok * DIM;
  const unsigned short* hrow = h_rd + (size_t)arow * DIM;

  // Per-lane B rows (output columns of the three gate blocks)
  const unsigned short* wx[3];
  const unsigned short* wh[3];
#pragma unroll
  for (int g = 0; g < 3; ++g) {
    wx[g] = wih_bf + (size_t)(g * DIM + n0 + r15) * DIM;
    wh[g] = whh_bf + (size_t)(g * DIM + n0 + r15) * DIM;
  }

  v8f accx[3] = {};
  v8f acch[3] = {};
  for (int kb = 0; kb < DIM; kb += 32) {
    v16bf ax = load_tile_bf16(xrow, kb, lane);
    v16bf ah = load_tile_bf16(hrow, kb, lane);
#pragma unroll
    for (int g = 0; g < 3; ++g) {
      accx[g] = WMMA_BF16(ax, load_tile_bf16(wx[g], kb, lane), accx[g]);
      acch[g] = WMMA_BF16(ah, load_tile_bf16(wh[g], kb, lane), acch[g]);
    }
  }

  // Gate math. C/D layout: VGPR v -> M = v + 8*half, N = lane&15.
  const int n = n0 + r15;
  const float bir = b_ih[n],           bhr = b_hh[n];
  const float biz = b_ih[DIM + n],     bhz = b_hh[DIM + n];
  const float bin = b_ih[2 * DIM + n], bhn = b_hh[2 * DIM + n];
#pragma unroll
  for (int v = 0; v < 8; ++v) {
    const int b = b0 + v + 8 * half;
    const float r  = sigf(accx[0][v] + bir + acch[0][v] + bhr);
    const float z  = sigf(accx[1][v] + biz + acch[1][v] + bhz);
    const float nn = tanhf(accx[2][v] + bin + r * (acch[2][v] + bhn));
    const size_t idx = (size_t)b * DIM + n;
    const float hnew = (1.0f - z) * nn + z * h_f32[idx];
    h_f32[idx] = hnew;
    const unsigned short hb = f2bf_bits(hnew);
    h_wr[idx] = hb;
    gru_bf[((size_t)b * T_FULL + t) * DIM + n] = hb;
  }
}

// ---------------------------------------------------------------------------
// Kernel 3: q2 = ht @ a2ᵀ  (WMMA), fused with w2[b,h] = out_w[h] * ht[b,h]
// (w2 folds c_global and the output projection into one per-batch vector:
//  out[b,t] = alpha[b,t] * <w2[b,:], gru_out[b,t,:]> + out_b)
// ---------------------------------------------------------------------------
__global__ void q2w2_kernel(const unsigned short* __restrict__ ht_bf,
                            const float* __restrict__ ht_f32,
                            const unsigned short* __restrict__ a2_bf,
                            const float* __restrict__ out_w,
                            float* __restrict__ q2,
                            float* __restrict__ w2) {
  const int lane = threadIdx.x;
  const int b0 = blockIdx.x * 16;
  const int n0 = blockIdx.y * 16;
  const int r15 = lane & 15;
  const int half = lane >> 4;

  const unsigned short* arow = ht_bf + (size_t)(b0 + r15) * DIM;
  const unsigned short* brow = a2_bf + (size_t)(n0 + r15) * DIM;

  v8f acc = {};
  for (int kb = 0; kb < DIM; kb += 32)
    acc = WMMA_BF16(load_tile_bf16(arow, kb, lane),
                    load_tile_bf16(brow, kb, lane), acc);

  const int n = n0 + r15;
  const float ow = out_w[n];
#pragma unroll
  for (int v = 0; v < 8; ++v) {
    const int b = b0 + v + 8 * half;
    const size_t idx = (size_t)b * DIM + n;
    q2[idx] = acc[v];
    w2[idx] = ow * ht_f32[idx];
  }
}

// ---------------------------------------------------------------------------
// Kernel 4: fused attention + output.
// Per 16-row (b,t) tile: q1 tile-row via WMMA (A hoisted into registers),
// mask+sigmoid+vt reduction -> alpha, then out = 0.1 + sigmoid(alpha*dot + b).
// ---------------------------------------------------------------------------
__global__ void attn_kernel(const unsigned short* __restrict__ gru_bf,
                            const unsigned short* __restrict__ a1_bf,
                            const float* __restrict__ q2,
                            const float* __restrict__ w2,
                            const float* __restrict__ vt_w,
                            const int* __restrict__ seq1,
                            const int* __restrict__ seq2,
                            const float* __restrict__ out_b,
                            float* __restrict__ out) {
  __shared__ float red[16][16];
  __shared__ float red2[16][2];
  __shared__ float alphaS[16];

  const int lane = threadIdx.x;
  const int m0 = blockIdx.x * 16;
  const int r15 = lane & 15;
  const int half = lane >> 4;

  // Hoist full A strip (16 rows x 256 K) for this tile: 8 operands.
  const unsigned short* arow = gru_bf + (size_t)(m0 + r15) * DIM;
  v16bf va[8];
#pragma unroll
  for (int i = 0; i < 8; ++i) va[i] = load_tile_bf16(arow, i * 32, lane);

  // Per-C-slot row metadata (batch index + mask)
  int bb[8];
  float mk[8];
#pragma unroll
  for (int v = 0; v < 8; ++v) {
    const int m = m0 + v + 8 * half;
    const int b = m / T_FULL;
    const int t = m - b * T_FULL;
    bb[v] = b;
    const int tk = (t < T1_LEN) ? seq1[b * T1_LEN + t] : seq2[b];
    mk[v] = (tk > 0) ? 1.0f : 0.0f;
  }

  float ap[8] = {0.f, 0.f, 0.f, 0.f, 0.f, 0.f, 0.f, 0.f};
  for (int nt = 0; nt < 16; ++nt) {
    const unsigned short* brow = a1_bf + (size_t)(nt * 16 + r15) * DIM;
    v8f c = {};
#pragma unroll
    for (int kb = 0; kb < 8; ++kb)
      c = WMMA_BF16(va[kb], load_tile_bf16(brow, kb * 32, lane), c);
    const int n = nt * 16 + r15;
    const float vt = vt_w[n];
#pragma unroll
    for (int v = 0; v < 8; ++v) {
      const float q = c[v] + mk[v] * q2[(size_t)bb[v] * DIM + n];
      ap[v] += vt * sigf(q);
    }
  }

  // Reduce alpha partials across lanes (16 n-columns each)
#pragma unroll
  for (int v = 0; v < 8; ++v) red[v + 8 * half][r15] = ap[v];
  __syncthreads();
  if (lane < 16) {
    float s = 0.f;
#pragma unroll
    for (int i = 0; i < 16; ++i) s += red[lane][i];
    alphaS[lane] = s;
  }
  __syncthreads();

  // dot(w2[b,:], gru_out[m,:]) — each half-lane-group handles 128 columns.
  const int m = m0 + r15;
  const int b = m / T_FULL;
  const u16x8* g = reinterpret_cast<const u16x8*>(gru_bf + (size_t)m * DIM + half * 128);
  const float4* wv = reinterpret_cast<const float4*>(w2 + (size_t)b * DIM + half * 128);
  float dot = 0.f;
#pragma unroll
  for (int i = 0; i < 16; i += 2) {
    u16x8 g0 = g[i], g1 = g[i + 1];
    float4 w0 = wv[2 * i + 0], w1 = wv[2 * i + 1];
    float4 w2v = wv[2 * i + 2], w3 = wv[2 * i + 3];
    dot += bfbits2f(g0[0]) * w0.x + bfbits2f(g0[1]) * w0.y +
           bfbits2f(g0[2]) * w0.z + bfbits2f(g0[3]) * w0.w;
    dot += bfbits2f(g0[4]) * w1.x + bfbits2f(g0[5]) * w1.y +
           bfbits2f(g0[6]) * w1.z + bfbits2f(g0[7]) * w1.w;
    dot += bfbits2f(g1[0]) * w2v.x + bfbits2f(g1[1]) * w2v.y +
           bfbits2f(g1[2]) * w2v.z + bfbits2f(g1[3]) * w2v.w;
    dot += bfbits2f(g1[4]) * w3.x + bfbits2f(g1[5]) * w3.y +
           bfbits2f(g1[6]) * w3.z + bfbits2f(g1[7]) * w3.w;
  }
  red2[r15][half] = dot;
  __syncthreads();

  if (lane < 16) {
    const float d = red2[lane][0] + red2[lane][1];
    const float o = alphaS[lane] * d + out_b[0];
    out[m0 + lane] = 0.1f + sigf(o);
  }
}

// ---------------------------------------------------------------------------
// Host launcher
// ---------------------------------------------------------------------------
extern "C" void kernel_launch(void* const* d_in, const int* in_sizes, int n_in,
                              void* d_out, int out_size, void* d_ws, size_t ws_size,
                              hipStream_t stream) {
  (void)in_sizes; (void)n_in; (void)out_size; (void)ws_size;

  const int*   seq1  = (const int*)d_in[0];
  const int*   seq2  = (const int*)d_in[1];
  const float* emb   = (const float*)d_in[2];
  const float* w_ih  = (const float*)d_in[3];
  const float* w_hh  = (const float*)d_in[4];
  const float* b_ih  = (const float*)d_in[5];
  const float* b_hh  = (const float*)d_in[6];
  const float* a1_w  = (const float*)d_in[7];
  const float* a2_w  = (const float*)d_in[8];
  const float* vt_w  = (const float*)d_in[9];
  const float* out_w = (const float*)d_in[10];
  const float* out_b = (const float*)d_in[11];
  float* out = (float*)d_out;

  // Workspace partition (all chunks 16B-aligned)
  char* p = (char*)d_ws;
  float* h_f32 = (float*)p;                 p += (size_t)BATCH * DIM * 4;      // 1 MB
  unsigned short* hb0 = (unsigned short*)p; p += (size_t)BATCH * DIM * 2;      // 0.5 MB
  unsigned short* hb1 = (unsigned short*)p; p += (size_t)BATCH * DIM * 2;      // 0.5 MB
  float* q2 = (float*)p;                    p += (size_t)BATCH * DIM * 4;      // 1 MB
  float* w2 = (float*)p;                    p += (size_t)BATCH * DIM * 4;      // 1 MB
  const int EMB_N = 20002 * DIM;
  unsigned short* emb_bf = (unsigned short*)p; p += (size_t)EMB_N * 2;         // 10.24 MB
  unsigned short* wih_bf = (unsigned short*)p; p += (size_t)3 * DIM * DIM * 2; // 384 KB
  unsigned short* whh_bf = (unsigned short*)p; p += (size_t)3 * DIM * DIM * 2; // 384 KB
  unsigned short* a1_bf  = (unsigned short*)p; p += (size_t)DIM * DIM * 2;     // 128 KB
  unsigned short* a2_bf  = (unsigned short*)p; p += (size_t)DIM * DIM * 2;     // 128 KB
  unsigned short* gru_bf = (unsigned short*)p; p += (size_t)M_ROWS * DIM * 2;  // 104 MB

  // One-time per-call conversions (weights stay L2-resident as bf16)
  {
    const int tb = 256;
    cvt_f32_bf16_kernel<<<(EMB_N + tb - 1) / tb, tb, 0, stream>>>(emb, emb_bf, EMB_N);
    cvt_f32_bf16_kernel<<<(3 * DIM * DIM + tb - 1) / tb, tb, 0, stream>>>(w_ih, wih_bf, 3 * DIM * DIM);
    cvt_f32_bf16_kernel<<<(3 * DIM * DIM + tb - 1) / tb, tb, 0, stream>>>(w_hh, whh_bf, 3 * DIM * DIM);
    cvt_f32_bf16_kernel<<<(DIM * DIM + tb - 1) / tb, tb, 0, stream>>>(a1_w, a1_bf, DIM * DIM);
    cvt_f32_bf16_kernel<<<(DIM * DIM + tb - 1) / tb, tb, 0, stream>>>(a2_w, a2_bf, DIM * DIM);
    init_state_kernel<<<(BATCH * DIM + tb - 1) / tb, tb, 0, stream>>>(h_f32, hb0);
  }

  // Serial GRU recurrence: one fused kernel per timestep, bf16 h ping-pong.
  const dim3 ggrid(BATCH / 16, DIM / 16);
  for (int t = 0; t < T_FULL; ++t) {
    const unsigned short* hr = (t & 1) ? hb1 : hb0;
    unsigned short*       hw = (t & 1) ? hb0 : hb1;
    gru_step_kernel<<<ggrid, 32, 0, stream>>>(seq1, seq2, emb_bf, wih_bf, whh_bf,
                                              b_ih, b_hh, h_f32, hr, hw, gru_bf, t);
  }
  // After t = 198 (even), final bf16 h lives in hb1.
  q2w2_kernel<<<ggrid, 32, 0, stream>>>(hb1, h_f32, a2_bf, out_w, q2, w2);

  attn_kernel<<<M_ROWS / 16, 32, 0, stream>>>(gru_bf, a1_bf, q2, w2, vt_w,
                                              seq1, seq2, out_b, out);
}
